// MoeLayer_38757784879510
// MI455X (gfx1250) — compile-verified, matrix operand order checked by reading, not run on hardware
//
#include <hip/hip_runtime.h>
#include <hip/hip_bf16.h>

// ---------------- constants (match reference shapes) ----------------
#define Tn   16384   // B*S tokens
#define Dn   512     // model dim
#define Hn   1365    // hidden dim
#define Hp   1408    // hidden padded to 88*16 (zero padded)
#define En   16      // experts
#define Kn   4       // top-k
#define Mtok 64      // tokens per workgroup (4 M-tiles of 16)
#define NW   8       // waves per workgroup

typedef __attribute__((ext_vector_type(16))) __bf16 v16bf;
typedef __attribute__((ext_vector_type(8)))  float  v8f;

static __device__ __forceinline__ v8f wmma_bf16(v16bf a, v16bf b, v8f c) {
    return __builtin_amdgcn_wmma_f32_16x16x32_bf16(
        /*neg_a=*/false, a, /*neg_b=*/false, b,
        /*c_mod=*/(short)0, c, /*reuse_a=*/false, /*reuse_b=*/false);
}

// ---------------- gating: logits, top-4, softmax, gather lists ----------------
__global__ __launch_bounds__(256)
void gate_kernel(const float* __restrict__ x, const float* __restrict__ gw,
                 const float* __restrict__ rb,
                 float* __restrict__ wsc, unsigned* __restrict__ idx,
                 int* __restrict__ cnt) {
    int t = blockIdx.x * 256 + threadIdx.x;
    if (t >= Tn) return;
    float logit[En];
#pragma unroll
    for (int e = 0; e < En; ++e) logit[e] = rb[e];
    const float* xr = x + (size_t)t * Dn;
    for (int d = 0; d < Dn; ++d) {
        float xv = xr[d];
        const float* g = gw + (size_t)d * En;
#pragma unroll
        for (int e = 0; e < En; ++e) logit[e] += xv * g[e];
    }
    // top-4 (ties -> lowest index, like jax top_k)
    int sel[Kn]; float sv[Kn]; unsigned used = 0;
#pragma unroll
    for (int k = 0; k < Kn; ++k) {
        int bi = 0; float bv = -3.0e38f;
#pragma unroll
        for (int e = 0; e < En; ++e)
            if (!((used >> e) & 1u) && logit[e] > bv) { bv = logit[e]; bi = e; }
        used |= 1u << bi; sel[k] = bi; sv[k] = bv;
    }
    float mx = sv[0];
    float p[Kn], s = 0.f;
#pragma unroll
    for (int k = 0; k < Kn; ++k) { p[k] = __expf(sv[k] - mx); s += p[k]; }
    float inv = 1.f / s;
#pragma unroll
    for (int k = 0; k < Kn; ++k) {
        wsc[t * Kn + k] = p[k] * inv;
        int e = sel[k];
        int pos = atomicAdd(&cnt[e], 1);
        idx[(size_t)e * Tn + pos] = ((unsigned)t << 2) | (unsigned)k;
    }
}

// ---------------- conversions into WMMA-friendly bf16 layouts ----------------
__global__ __launch_bounds__(256)
void conv_x_kernel(const float* __restrict__ x, __bf16* __restrict__ xb) {
    size_t i = ((size_t)blockIdx.x * 256 + threadIdx.x) * 4;
    if (i >= (size_t)Tn * Dn) return;
    const float4 v = *(const float4*)(x + i);
    xb[i + 0] = (__bf16)v.x; xb[i + 1] = (__bf16)v.y;
    xb[i + 2] = (__bf16)v.z; xb[i + 3] = (__bf16)v.w;
}

// w1t/w2t: [E][Hp][Dn], w1t[e][h][d] = w1[e][d][h] (zero for h>=Hn)
__global__ __launch_bounds__(256)
void conv_w12_kernel(const float* __restrict__ w1, const float* __restrict__ w2,
                     __bf16* __restrict__ w1t, __bf16* __restrict__ w2t) {
    size_t i = (size_t)blockIdx.x * 256 + threadIdx.x;
    if (i >= (size_t)En * Hp * Dn) return;
    size_t e = i / ((size_t)Hp * Dn);
    size_t r = i % ((size_t)Hp * Dn);
    size_t h = r / Dn, d = r % Dn;
    float a = 0.f, b = 0.f;
    if (h < Hn) {
        size_t src = ((size_t)e * Dn + d) * Hn + h;
        a = w1[src]; b = w2[src];
    }
    w1t[i] = (__bf16)a; w2t[i] = (__bf16)b;
}

// w3t: [E][Dn][Hp], w3t[e][d][h] = w3[e][h][d] (zero for h>=Hn)
__global__ __launch_bounds__(256)
void conv_w3_kernel(const float* __restrict__ w3, __bf16* __restrict__ w3t) {
    size_t i = (size_t)blockIdx.x * 256 + threadIdx.x;
    if (i >= (size_t)En * Dn * Hp) return;
    size_t e = i / ((size_t)Dn * Hp);
    size_t r = i % ((size_t)Dn * Hp);
    size_t d = r / Hp, h = r % Hp;
    float a = 0.f;
    if (h < Hn) a = w3[((size_t)e * Hn + h) * Dn + d];
    w3t[i] = (__bf16)a;
}

// shared expert weights: sw1t/sw2t [Hp][Dn]; sw3t [Dn][Hp]
__global__ __launch_bounds__(256)
void conv_sw_kernel(const float* __restrict__ sw1, const float* __restrict__ sw2,
                    const float* __restrict__ sw3,
                    __bf16* __restrict__ sw1t, __bf16* __restrict__ sw2t,
                    __bf16* __restrict__ sw3t) {
    size_t i = (size_t)blockIdx.x * 256 + threadIdx.x;
    if (i >= (size_t)Hp * Dn) return;
    size_t h = i / Dn, d = i % Dn;
    float a = 0.f, b = 0.f, c = 0.f;
    if (h < Hn) {
        a = sw1[d * Hn + h];      // sw1 is [D][H]
        b = sw2[d * Hn + h];
        c = sw3[h * Dn + d];      // sw3 is [H][D]
    }
    sw1t[h * Dn + d] = (__bf16)a;
    sw2t[h * Dn + d] = (__bf16)b;
    sw3t[d * Hp + h] = (__bf16)c;
}

// ---------------- main WMMA tile kernel (experts + shared) ----------------
// One block = 64 (gathered) tokens = 4 M-tiles; 8 waves of 32.
// Each B-fragment load feeds 4 (GEMM2) or 8 (GEMM1, w1&w2) WMMAs -> 4x less
// weight traffic from L2 than an M=16 tile.
// GEMM1: [64 x 512] @ [512 x 1408] (w1 & w2) -> swiglu -> h in LDS (bf16)
// GEMM2: [64 x 1408] @ [1408 x 512] -> weighted scatter to contrib / d_out
__global__ __launch_bounds__(256)
void moe_tile_kernel(const __bf16* __restrict__ xb,
                     const __bf16* __restrict__ wA,   // [E][Hp][Dn]
                     const __bf16* __restrict__ wB,   // [E][Hp][Dn]
                     const __bf16* __restrict__ wC,   // [E][Dn][Hp]
                     const unsigned* __restrict__ idx,
                     const int* __restrict__ cnt,
                     const float* __restrict__ wsc,
                     float* __restrict__ dst,
                     int is_shared) {
    extern __shared__ char smem[];
    __bf16* xs = (__bf16*)smem;                              // [Mtok][Dn]  64 KB
    __bf16* hs = (__bf16*)(smem + (size_t)Mtok * Dn * 2);    // [Mtok][Hp] 176 KB
    __shared__ float sc_s[Mtok];
    __shared__ int   dst_s[Mtok];
    __shared__ int   tok_s[Mtok];

    const int e    = blockIdx.y;
    const int tile = blockIdx.x;
    const int tid  = threadIdx.x;

    if (!is_shared) {
        if (tile * Mtok >= cnt[e]) return;   // uniform across block
    }

    if (tid < Mtok) {
        if (is_shared) {
            int t = tile * Mtok + tid;
            tok_s[tid] = t; dst_s[tid] = t; sc_s[tid] = 1.f;
        } else {
            unsigned ent = idx[(size_t)e * Tn + tile * Mtok + tid];
            if (ent == 0xFFFFFFFFu) {
                tok_s[tid] = -1; dst_s[tid] = -1; sc_s[tid] = 0.f;
            } else {
                int t = (int)(ent >> 2), k = (int)(ent & 3u);
                tok_s[tid] = t;
                dst_s[tid] = t * Kn + k;
                sc_s[tid]  = wsc[t * Kn + k];
            }
        }
    }
    __syncthreads();

    // stage x tile into LDS: 4 threads per row, 128 bf16 (256B) per thread
    {
        int r = tid >> 2;
        int cseg = (tid & 3) * 128;
        int tok = tok_s[r];
        uint4* dp = (uint4*)(&xs[r * Dn + cseg]);
        if (tok >= 0) {
            const uint4* sp = (const uint4*)(xb + (size_t)tok * Dn + cseg);
#pragma unroll
            for (int i = 0; i < 16; ++i) dp[i] = sp[i];
        } else {
            uint4 z = {0u, 0u, 0u, 0u};
#pragma unroll
            for (int i = 0; i < 16; ++i) dp[i] = z;
        }
    }
    __syncthreads();

    const int wave  = tid >> 5;
    const int lane  = tid & 31;
    const int rowA  = lane & 15;            // A frag: M row / B frag: N col
    const int ksel  = (lane >> 4) << 4;     // K sub-chunk select (0 or 16)
    const int mbase = (lane >> 4) << 3;     // C/D frag: M base (0 or 8)

    const __bf16* wAe = wA + (size_t)e * Hp * Dn;
    const __bf16* wBe = wB + (size_t)e * Hp * Dn;
    const __bf16* wCe = wC + (size_t)e * Dn * Hp;

    // ---- GEMM1 + fused SwiGLU: each wave owns 11 of the 88 H-tiles ----
    for (int t1 = 0; t1 < 11; ++t1) {
        int nt = wave * 11 + t1;
        v8f acc1[4] = {}; v8f acc2[4] = {};
        const __bf16* b1p = wAe + (size_t)(nt * 16 + rowA) * Dn + ksel;
        const __bf16* b2p = wBe + (size_t)(nt * 16 + rowA) * Dn + ksel;
        // prefetch next H-tile's weight rows for this lane
        __builtin_prefetch(b1p + (size_t)16 * Dn, 0, 1);
        __builtin_prefetch(b2p + (size_t)16 * Dn, 0, 1);
#pragma unroll 4
        for (int kb = 0; kb < 16; ++kb) {
            v16bf b1 = *(const v16bf*)(b1p + kb * 32);
            v16bf b2 = *(const v16bf*)(b2p + kb * 32);
#pragma unroll
            for (int mt = 0; mt < 4; ++mt) {
                v16bf a = *(const v16bf*)(&xs[(mt * 16 + rowA) * Dn + kb * 32 + ksel]);
                acc1[mt] = wmma_bf16(a, b1, acc1[mt]);
                acc2[mt] = wmma_bf16(a, b2, acc2[mt]);
            }
        }
        int ncol = (lane & 15) + nt * 16;
#pragma unroll
        for (int mt = 0; mt < 4; ++mt) {
#pragma unroll
            for (int i = 0; i < 8; ++i) {
                float a1 = acc1[mt][i], a2 = acc2[mt][i];
                float g = a1 / (1.f + __expf(-a1)) * a2;   // swish(a1) * a2
                hs[(mt * 16 + mbase + i) * Hp + ncol] = (__bf16)g;
            }
        }
    }
    __syncthreads();

    // ---- GEMM2: each wave owns 4 of the 32 D-tiles ----
    for (int t2 = 0; t2 < 4; ++t2) {
        int nt = wave * 4 + t2;
        v8f acc[4] = {};
        const __bf16* b3p = wCe + (size_t)(nt * 16 + rowA) * Hp + ksel;
        __builtin_prefetch(b3p + (size_t)16 * Hp, 0, 1);
#pragma unroll 4
        for (int kb = 0; kb < 44; ++kb) {
            v16bf b = *(const v16bf*)(b3p + kb * 32);
#pragma unroll
            for (int mt = 0; mt < 4; ++mt) {
                v16bf a = *(const v16bf*)(&hs[(mt * 16 + rowA) * Hp + kb * 32 + ksel]);
                acc[mt] = wmma_bf16(a, b, acc[mt]);
            }
        }
        int ncol = (lane & 15) + nt * 16;
#pragma unroll
        for (int mt = 0; mt < 4; ++mt) {
#pragma unroll
            for (int i = 0; i < 8; ++i) {
                int m = mt * 16 + mbase + i;
                int dr = dst_s[m];
                if (dr >= 0) dst[(size_t)dr * Dn + ncol] = acc[mt][i] * sc_s[m];
            }
        }
    }
}

// ---------------- combine: out[t] += sum_k contrib[t][k] ----------------
__global__ __launch_bounds__(256)
void combine_kernel(float* __restrict__ out, const float* __restrict__ contrib) {
    size_t i4 = (size_t)blockIdx.x * 256 + threadIdx.x;     // float4 index
    if (i4 >= (size_t)Tn * Dn / 4) return;
    size_t t  = i4 / (Dn / 4);
    size_t dv = i4 % (Dn / 4);
    float4 o = ((const float4*)out)[i4];
#pragma unroll
    for (int k = 0; k < Kn; ++k) {
        float4 c = ((const float4*)contrib)[(t * Kn + k) * (Dn / 4) + dv];
        o.x += c.x; o.y += c.y; o.z += c.z; o.w += c.w;
    }
    ((float4*)out)[i4] = o;
}

// ---------------- host side ----------------
static inline size_t align_up(size_t v, size_t a) { return (v + a - 1) & ~(a - 1); }

extern "C" void kernel_launch(void* const* d_in, const int* in_sizes, int n_in,
                              void* d_out, int out_size, void* d_ws, size_t ws_size,
                              hipStream_t stream) {
    const float* x   = (const float*)d_in[0];
    const float* gw  = (const float*)d_in[1];
    const float* rb  = (const float*)d_in[2];
    const float* w1  = (const float*)d_in[3];
    const float* w2  = (const float*)d_in[4];
    const float* w3  = (const float*)d_in[5];
    const float* sw1 = (const float*)d_in[6];
    const float* sw2 = (const float*)d_in[7];
    const float* sw3 = (const float*)d_in[8];
    float* out = (float*)d_out;

    // workspace carve-up
    char* p = (char*)d_ws;
    size_t off = 0;
    auto carve = [&](size_t bytes) { char* r = p + off; off = align_up(off + bytes, 256); return r; };

    int*      cnt     = (int*)     carve(sizeof(int) * En);
    unsigned* idx     = (unsigned*)carve(sizeof(unsigned) * (size_t)En * Tn);
    float*    wsc     = (float*)   carve(sizeof(float) * (size_t)Tn * Kn);
    __bf16*   xb      = (__bf16*)  carve(sizeof(__bf16) * (size_t)Tn * Dn);
    __bf16*   w1t     = (__bf16*)  carve(sizeof(__bf16) * (size_t)En * Hp * Dn);
    __bf16*   w2t     = (__bf16*)  carve(sizeof(__bf16) * (size_t)En * Hp * Dn);
    __bf16*   w3t     = (__bf16*)  carve(sizeof(__bf16) * (size_t)En * Dn * Hp);
    __bf16*   sw1t    = (__bf16*)  carve(sizeof(__bf16) * (size_t)Hp * Dn);
    __bf16*   sw2t    = (__bf16*)  carve(sizeof(__bf16) * (size_t)Hp * Dn);
    __bf16*   sw3t    = (__bf16*)  carve(sizeof(__bf16) * (size_t)Dn * Hp);
    float*    contrib = (float*)   carve(sizeof(float) * (size_t)Tn * Kn * Dn);
    (void)ws_size;

    // reset gather state
    hipMemsetAsync(cnt, 0, sizeof(int) * En, stream);
    hipMemsetAsync(idx, 0xFF, sizeof(unsigned) * (size_t)En * Tn, stream);

    // gating
    gate_kernel<<<Tn / 256, 256, 0, stream>>>(x, gw, rb, wsc, idx, cnt);

    // conversions
    conv_x_kernel<<<(unsigned)(((size_t)Tn * Dn / 4 + 255) / 256), 256, 0, stream>>>(x, xb);
    conv_w12_kernel<<<(unsigned)(((size_t)En * Hp * Dn + 255) / 256), 256, 0, stream>>>(w1, w2, w1t, w2t);
    conv_w3_kernel<<<(unsigned)(((size_t)En * Dn * Hp + 255) / 256), 256, 0, stream>>>(w3, w3t);
    conv_sw_kernel<<<(unsigned)(((size_t)Hp * Dn + 255) / 256), 256, 0, stream>>>(sw1, sw2, sw3, sw1t, sw2t, sw3t);

    const size_t smem_bytes = ((size_t)Mtok * Dn + (size_t)Mtok * Hp) * sizeof(__bf16); // 240 KB

    // shared expert: writes d_out directly (scale 1, identity gather)
    {
        dim3 grid(Tn / Mtok, 1);
        moe_tile_kernel<<<grid, 256, smem_bytes, stream>>>(xb, sw1t, sw2t, sw3t,
                                                           nullptr, nullptr, nullptr,
                                                           out, /*is_shared=*/1);
    }
    // routed experts: weighted rows into conflict-free contrib[t][k][D]
    {
        dim3 grid(Tn / Mtok, En);
        moe_tile_kernel<<<grid, 256, smem_bytes, stream>>>(xb, w1t, w2t, w3t,
                                                           idx, cnt, wsc,
                                                           contrib, /*is_shared=*/0);
    }
    // out += sum_k contrib
    combine_kernel<<<(unsigned)(((size_t)Tn * Dn / 4 + 255) / 256), 256, 0, stream>>>(out, contrib);
}
